// HybridGINSchNet_35167192220006
// MI455X (gfx1250) — matrix operand order repaired and checked by reading, not compiled, not run on hardware
//
#include <hip/hip_runtime.h>
#include <math.h>

#define NN      16384
#define NGRAPH  512
#define E2C     65536
#define E3C     393216
#define EMB     300
#define EMB2    600
#define SCH     128
#define NGAUSS  50

typedef __attribute__((ext_vector_type(16))) __bf16 v16bf;
typedef __attribute__((ext_vector_type(8)))  float  v8f;

union FragBF { v16bf v; unsigned int u[8]; };

#if defined(__has_builtin)
#if __has_builtin(__builtin_amdgcn_global_load_async_to_lds_b32) && \
    __has_builtin(__builtin_amdgcn_s_wait_asynccnt)
#define HAVE_ASYNC_LDS 1
#endif
#endif
#ifndef HAVE_ASYNC_LDS
#define HAVE_ASYNC_LDS 0
#endif

#if HAVE_ASYNC_LDS
typedef __attribute__((address_space(1))) int* as1_iptr;
typedef __attribute__((address_space(3))) int* as3_iptr;
#endif

__device__ __forceinline__ unsigned short f2bf(float f) {
  unsigned int u = __float_as_uint(f);
  u += 0x7fffu + ((u >> 16) & 1u);          // round-to-nearest-even
  return (unsigned short)(u >> 16);
}

__device__ __forceinline__ float sspf(float x) {  // shifted softplus
  float sp = (x > 15.0f) ? x : log1pf(__expf(x));
  return sp - 0.69314718056f;
}

// ---------------------------------------------------------------------------
// Fused bf16 WMMA GEMM:  out = act( (A@B + bias) [*gamma+beta] [*rowscale] ) [+residual]
// A: [M,K] bf16 row-major (lda), B: [K,N] bf16 row-major (ldb).
// Block tile 128x64, 8 waves (4x2), each wave 2x2 WMMA 16x16 tiles, K-step 32.
// Interior tiles staged with unguarded u32 (async-to-LDS when available);
// boundary tiles fall back to guarded u16 staging.
// ---------------------------------------------------------------------------
#define BM 128
#define BN 64
#define BK 32
#define LDA_S 40   // even -> 32-bit aligned LDS accesses
#define LDB_S 40

__global__ __launch_bounds__(256) void k_gemm(
    const unsigned short* __restrict__ A, int lda,
    const unsigned short* __restrict__ B, int ldb,
    int M, int N, int K,
    const float* __restrict__ bias,
    const float* __restrict__ gamma,
    const float* __restrict__ beta,
    const float* __restrict__ rowscale,
    const float* __restrict__ residual, int ldres,
    int act,
    float* __restrict__ outF, int ldof,
    unsigned short* __restrict__ outB, int ldob)
{
  __shared__ __align__(16) unsigned short sA[BM * LDA_S];
  __shared__ __align__(16) unsigned short sBt[BN * LDB_S];

  const int tid   = threadIdx.x;
  const int lane  = tid & 31;
  const int wave  = tid >> 5;
  const int wm    = wave & 3;     // 4 waves along M
  const int wn    = wave >> 2;    // 2 waves along N
  const int bm    = blockIdx.x * BM;
  const int bn    = blockIdx.y * BN;
  const int row16 = lane & 15;
  const int khalf = lane >> 4;
  const bool iA   = (bm + BM <= M);   // interior along M
  const bool iB   = (bn + BN <= N);   // interior along N

  v8f acc[2][2] = {};

  for (int k0 = 0; k0 < K; k0 += BK) {
    const bool fullK = (k0 + BK <= K);
#if HAVE_ASYNC_LDS
    bool usedAsync = false;
#endif
    // ---- stage A tile: 128x32 ----
    if (iA && fullK) {
#if HAVE_ASYNC_LDS
      usedAsync = true;
#pragma unroll
      for (int i = 0; i < 8; ++i) {               // 2048 dwords / 256 thr = 8
        int idx = i * 256 + tid;
        int r = idx >> 4, c2 = (idx & 15) << 1;   // 16 dwords per 32-elem row
        __builtin_amdgcn_global_load_async_to_lds_b32(
            (as1_iptr)(A + (size_t)(bm + r) * lda + k0 + c2),
            (as3_iptr)(&sA[r * LDA_S + c2]), 0, 0);
      }
#else
#pragma unroll
      for (int i = 0; i < 8; ++i) {
        int idx = i * 256 + tid;
        int r = idx >> 4, c2 = (idx & 15) << 1;
        *(unsigned int*)&sA[r * LDA_S + c2] =
            *(const unsigned int*)(A + (size_t)(bm + r) * lda + k0 + c2);
      }
#endif
    } else {
#pragma unroll
      for (int i = 0; i < 16; ++i) {              // guarded boundary path
        int idx = i * 256 + tid;
        int r = idx >> 5, c = idx & 31;
        int gr = bm + r, gc = k0 + c;
        unsigned short v = 0;
        if (gr < M && gc < K) v = A[(size_t)gr * lda + gc];
        sA[r * LDA_S + c] = v;
      }
    }
    // ---- stage B tile transposed: [N=64][K=32] ----
    if (iB && fullK) {
#pragma unroll
      for (int i = 0; i < 4; ++i) {               // 1024 dwords / 256 thr = 4
        int idx = i * 256 + tid;
        int kk = idx >> 5, n2 = (idx & 31) << 1;
        unsigned int v = *(const unsigned int*)(B + (size_t)(k0 + kk) * ldb + bn + n2);
        sBt[n2 * LDB_S + kk]       = (unsigned short)v;
        sBt[(n2 + 1) * LDB_S + kk] = (unsigned short)(v >> 16);
      }
    } else {
#pragma unroll
      for (int i = 0; i < 8; ++i) {
        int idx = i * 256 + tid;
        int kk = idx >> 6, nn = idx & 63;
        int gk = k0 + kk, gn = bn + nn;
        unsigned short v = 0;
        if (gk < K && gn < N) v = B[(size_t)gk * ldb + gn];
        sBt[nn * LDB_S + kk] = v;
      }
    }
    // ---- prefetch next K-tile while this one drains ----
    if (iA && (k0 + BK < K))
      __builtin_prefetch(A + (size_t)(bm + (tid & 127)) * lda + k0 + BK, 0, 3);
    if (iB && (k0 + BK + lane < K) && wave == 0)
      __builtin_prefetch(B + (size_t)(k0 + BK + lane) * ldb + bn, 0, 3);
#if HAVE_ASYNC_LDS
    if (usedAsync) __builtin_amdgcn_s_wait_asynccnt(0);
#endif
    __syncthreads();

    // ---- assemble fragments, contiguous K = 16*khalf + e in A and B (wave32) ----
    FragBF bfr[2];
#pragma unroll
    for (int ni = 0; ni < 2; ++ni) {
      int n = wn * 32 + ni * 16 + row16;           // lane -> column
      const unsigned int* pb = (const unsigned int*)&sBt[n * LDB_S];
#pragma unroll
      for (int v = 0; v < 8; ++v) bfr[ni].u[v] = pb[8 * khalf + v];  // K = e + 16*khalf
    }
    FragBF afr[2];
#pragma unroll
    for (int mi = 0; mi < 2; ++mi) {
      int r = wm * 32 + mi * 16 + row16;           // lane -> row
      const unsigned int* pa = (const unsigned int*)&sA[r * LDA_S];
#pragma unroll
      for (int v = 0; v < 8; ++v) {
        int kp = 2 * v + 16 * khalf;  // K-pair base per VGPR
        afr[mi].u[v] = pa[kp >> 1];
      }
    }
#pragma unroll
    for (int mi = 0; mi < 2; ++mi)
#pragma unroll
      for (int ni = 0; ni < 2; ++ni)
        acc[mi][ni] = __builtin_amdgcn_wmma_f32_16x16x32_bf16(
            false, afr[mi].v, false, bfr[ni].v, (short)0, acc[mi][ni], false, false);
    __syncthreads();
  }

  // ---- fused epilogue (C/D layout: N=lane&15, M = v + 8*khalf per VGPR v) ----
#pragma unroll
  for (int mi = 0; mi < 2; ++mi) {
#pragma unroll
    for (int ni = 0; ni < 2; ++ni) {
      int n = bn + wn * 32 + ni * 16 + row16;
      if (n >= N) continue;
      float bval  = bias  ? bias[n]  : 0.0f;
      float gval  = gamma ? gamma[n] : 1.0f;
      float btval = (gamma && beta) ? beta[n] : 0.0f;
#pragma unroll
      for (int v = 0; v < 8; ++v) {
        int m = bm + wm * 32 + mi * 16 + 8 * khalf + v;
        if (m >= M) continue;
        float y = acc[mi][ni][v] + bval;
        if (gamma)    y = y * gval + btval;
        if (rowscale) y *= rowscale[m];
        if (act == 1)      y = fmaxf(y, 0.0f);
        else if (act == 2) y = sspf(y);
        if (residual) y += residual[(size_t)m * ldres + n];
        if (outF) outF[(size_t)m * ldof + n] = y;
        if (outB) outB[(size_t)m * ldob + n] = f2bf(y);
      }
    }
  }
}

// ---------------------------------------------------------------------------
// Elementwise / gather / scatter kernels (fp32, bandwidth-bound path)
// ---------------------------------------------------------------------------
__global__ void k_zero(float* p, long long n) {
  long long i = (long long)blockIdx.x * blockDim.x + threadIdx.x;
  if (i < n) p[i] = 0.0f;
}
__global__ void k_copy(const float* a, float* b, long long n) {
  long long i = (long long)blockIdx.x * blockDim.x + threadIdx.x;
  if (i < n) b[i] = a[i];
}
__global__ void k_f32bf(const float* a, unsigned short* b, long long n) {
  long long i = (long long)blockIdx.x * blockDim.x + threadIdx.x;
  if (i < n) b[i] = f2bf(a[i]);
}
__global__ void k_counts(const int* batch, float* cnt) {
  int n = blockIdx.x * blockDim.x + threadIdx.x;
  if (n < NN) atomicAdd(&cnt[batch[n]], 1.0f);
}
__global__ void k_invcnt(float* cnt) {
  int g = blockIdx.x * blockDim.x + threadIdx.x;
  if (g < NGRAPH) cnt[g] = 1.0f / fmaxf(cnt[g], 1.0f);
}
__global__ void k_gin_embed(const int* x2d, const float* e1, const float* e2, float* h) {
  long long i = (long long)blockIdx.x * blockDim.x + threadIdx.x;
  if (i >= (long long)NN * EMB) return;
  int n = (int)(i / EMB), f = (int)(i % EMB);
  h[i] = e1[(size_t)x2d[n * 2 + 0] * EMB + f] + e2[(size_t)x2d[n * 2 + 1] * EMB + f];
}
__global__ void k_bcast_vn(const float* vn_init, float* vn) {
  long long i = (long long)blockIdx.x * blockDim.x + threadIdx.x;
  if (i < (long long)NGRAPH * EMB) vn[i] = vn_init[i % EMB];
}
__global__ void k_addvn(const float* h, const float* vn, const int* batch, float* h_in) {
  long long i = (long long)blockIdx.x * blockDim.x + threadIdx.x;
  if (i >= (long long)NN * EMB) return;
  int n = (int)(i / EMB), f = (int)(i % EMB);
  h_in[i] = h[i] + vn[(size_t)batch[n] * EMB + f];
}
__global__ void k_gin_msg(const int* __restrict__ src, const int* __restrict__ dst,
                          const int* __restrict__ eattr,
                          const float* __restrict__ bond1, const float* __restrict__ bond2,
                          const float* __restrict__ h_in, float* __restrict__ agg) {
  int e = blockIdx.x;
  int s = src[e], d = dst[e];
  int a0 = eattr[e * 2 + 0], a1 = eattr[e * 2 + 1];
  for (int f = threadIdx.x; f < EMB; f += blockDim.x) {
    float m = h_in[(size_t)s * EMB + f] + bond1[(size_t)a0 * EMB + f] + bond2[(size_t)a1 * EMB + f];
    atomicAdd(&agg[(size_t)d * EMB + f], fmaxf(m, 0.0f));
  }
}
__global__ void k_add2bf(const float* a, const float* b, unsigned short* o, long long n) {
  long long i = (long long)blockIdx.x * blockDim.x + threadIdx.x;
  if (i < n) o[i] = f2bf(a[i] + b[i]);
}
__global__ void k_segsum(const float* x, const int* batch, float* out, int F) {
  long long i = (long long)blockIdx.x * blockDim.x + threadIdx.x;
  if (i >= (long long)NN * F) return;
  int n = (int)(i / F), f = (int)(i % F);
  atomicAdd(&out[(size_t)batch[n] * F + f], x[i]);
}
__global__ void k_rowscale(float* x, const float* s, int rows, int F) {
  long long i = (long long)blockIdx.x * blockDim.x + threadIdx.x;
  if (i < (long long)rows * F) x[i] *= s[i / F];
}
__global__ void k_rbf(const int* __restrict__ src, const int* __restrict__ dst,
                      const float* __restrict__ pos,
                      unsigned short* __restrict__ rbf, float* __restrict__ C) {
  int e = blockIdx.x * blockDim.x + threadIdx.x;
  if (e >= E3C) return;
  int s = src[e], d = dst[e];
  float dx = pos[d * 3 + 0] - pos[s * 3 + 0];
  float dy = pos[d * 3 + 1] - pos[s * 3 + 1];
  float dz = pos[d * 3 + 2] - pos[s * 3 + 2];
  float dist = sqrtf(dx * dx + dy * dy + dz * dz);
  const float delta = 10.0f / (NGAUSS - 1);
  const float coeff = -0.5f / (delta * delta);
  for (int g = 0; g < NGAUSS; ++g) {
    float t = dist - (float)g * delta;
    rbf[(size_t)e * NGAUSS + g] = f2bf(__expf(coeff * t * t));
  }
  C[e] = 0.5f * (__cosf(dist * 3.14159265358979f / 10.0f) + 1.0f);
}
__global__ void k_sch_embed(const int* z, const float* emb, float* hs, unsigned short* hsbf) {
  long long i = (long long)blockIdx.x * blockDim.x + threadIdx.x;
  if (i >= (long long)NN * SCH) return;
  int n = (int)(i >> 7), f = (int)(i & 127);
  float v = emb[(size_t)z[n] * SCH + f];
  hs[i] = v; hsbf[i] = f2bf(v);
}
__global__ void k_cfconv(const int* __restrict__ src, const int* __restrict__ dst,
                         const float* __restrict__ X1, const float* __restrict__ Wf,
                         float* __restrict__ agg) {
  int e = blockIdx.x;
  int s = src[e], d = dst[e];
  int f = threadIdx.x;  // 128 threads
  atomicAdd(&agg[(size_t)d * SCH + f], X1[(size_t)s * SCH + f] * Wf[(size_t)e * SCH + f]);
}
__global__ void k_gate(const float* gt, const float* gW2, const float* gb2, float* g) {
  int b = blockIdx.x * blockDim.x + threadIdx.x;
  if (b >= NGRAPH) return;
  float s = gb2[0];
  for (int i = 0; i < 128; ++i) s += gt[(size_t)b * 128 + i] * gW2[i];
  g[b] = 1.0f / (1.0f + __expf(-s));
}
__global__ void k_fuse(const float* g, const float* cat, unsigned short* hfbf) {
  long long i = (long long)blockIdx.x * blockDim.x + threadIdx.x;
  if (i >= (long long)NGRAPH * EMB) return;
  int b = (int)(i / EMB), f = (int)(i % EMB);
  float gg = g[b];
  float h2p = cat[(size_t)b * EMB2 + f];
  float h3p = cat[(size_t)b * EMB2 + EMB + f];
  hfbf[i] = f2bf(gg * h2p + (1.0f - gg) * h3p);
}

// ---------------------------------------------------------------------------
// Host-side orchestration
// ---------------------------------------------------------------------------
static inline unsigned blk(long long n) { return (unsigned)((n + 255) / 256); }

static void gemm(hipStream_t st, const unsigned short* A, int lda,
                 const unsigned short* B, int ldb, int M, int N, int K,
                 const float* bias, const float* gamma, const float* beta,
                 const float* rowscale, const float* residual, int ldres, int act,
                 float* outF, int ldof, unsigned short* outB, int ldob) {
  dim3 grid((unsigned)((M + BM - 1) / BM), (unsigned)((N + BN - 1) / BN));
  k_gemm<<<grid, 256, 0, st>>>(A, lda, B, ldb, M, N, K, bias, gamma, beta,
                               rowscale, residual, ldres, act, outF, ldof, outB, ldob);
}
static void tobf(hipStream_t st, const float* a, unsigned short* b, long long n) {
  k_f32bf<<<blk(n), 256, 0, st>>>(a, b, n);
}
static void zerof(hipStream_t st, float* p, long long n) {
  k_zero<<<blk(n), 256, 0, st>>>(p, n);
}

extern "C" void kernel_launch(void* const* d_in, const int* in_sizes, int n_in,
                              void* d_out, int out_size, void* d_ws, size_t ws_size,
                              hipStream_t stream) {
  const int*   x2d       = (const int*)d_in[0];
  const int*   ei2       = (const int*)d_in[1];
  const int*   ea2       = (const int*)d_in[2];
  const int*   batch     = (const int*)d_in[3];
  const int*   z         = (const int*)d_in[4];
  const float* pos       = (const float*)d_in[5];
  const int*   ei3       = (const int*)d_in[6];
  const float* atom_emb1 = (const float*)d_in[7];
  const float* atom_emb2 = (const float*)d_in[8];
  const float* vn_init   = (const float*)d_in[9];
  const float* gin_W1    = (const float*)d_in[10];
  const float* gin_b1    = (const float*)d_in[11];
  const float* gin_W2    = (const float*)d_in[12];
  const float* gin_b2    = (const float*)d_in[13];
  const float* gin_bond1 = (const float*)d_in[14];
  const float* gin_bond2 = (const float*)d_in[15];
  const float* gin_gamma = (const float*)d_in[16];
  const float* gin_beta  = (const float*)d_in[17];
  const float* vn_W1     = (const float*)d_in[18];
  const float* vn_b1     = (const float*)d_in[19];
  const float* vn_W2     = (const float*)d_in[20];
  const float* vn_b2     = (const float*)d_in[21];
  const float* sch_emb   = (const float*)d_in[22];
  const float* s_mW1     = (const float*)d_in[23];
  const float* s_mb1     = (const float*)d_in[24];
  const float* s_mW2     = (const float*)d_in[25];
  const float* s_mb2     = (const float*)d_in[26];
  const float* s_lin1W   = (const float*)d_in[27];
  const float* s_lin2W   = (const float*)d_in[28];
  const float* s_lin2b   = (const float*)d_in[29];
  const float* s_linW    = (const float*)d_in[30];
  const float* s_linb    = (const float*)d_in[31];
  const float* p2W       = (const float*)d_in[32];
  const float* p2b       = (const float*)d_in[33];
  const float* p3W       = (const float*)d_in[34];
  const float* p3b       = (const float*)d_in[35];
  const float* gW1       = (const float*)d_in[36];
  const float* gb1       = (const float*)d_in[37];
  const float* gW2       = (const float*)d_in[38];
  const float* gb2       = (const float*)d_in[39];
  const float* cW1       = (const float*)d_in[40];
  const float* cb1       = (const float*)d_in[41];
  const float* cW2       = (const float*)d_in[42];
  const float* cb2       = (const float*)d_in[43];

  const int* src2 = ei2, *dst2 = ei2 + E2C;
  const int* src3 = ei3, *dst3 = ei3 + E3C;

  // ---- workspace bump allocator ----
  char* ws = (char*)d_ws;
  size_t off = 0;
  auto alloc = [&](size_t bytes) -> void* {
    void* p = ws + off;
    off = (off + bytes + 255) & ~(size_t)255;
    return p;
  };
  float* cntinv  = (float*)alloc((size_t)NGRAPH * 4);
  float* h       = (float*)alloc((size_t)NN * EMB * 4);
  float* h_in    = (float*)alloc((size_t)NN * EMB * 4);
  float* agg     = (float*)alloc((size_t)NN * EMB * 4);            // reused (300 & 128 wide)
  unsigned short* xbfA = (unsigned short*)alloc((size_t)NN * EMB2 * 2);
  unsigned short* xbfB = (unsigned short*)alloc((size_t)NN * EMB2 * 2);
  unsigned short* wbf  = (unsigned short*)alloc((size_t)EMB * EMB2 * 2);  // weight scratch
  float* vn      = (float*)alloc((size_t)NGRAPH * EMB * 4);
  float* vn_tmp  = (float*)alloc((size_t)NGRAPH * EMB * 4);
  unsigned short* vnbfA = (unsigned short*)alloc((size_t)NGRAPH * EMB * 2);
  unsigned short* vnbfB = (unsigned short*)alloc((size_t)NGRAPH * EMB2 * 2);
  float* h2d     = (float*)alloc((size_t)NGRAPH * EMB * 4);
  float* h3d     = (float*)alloc((size_t)NGRAPH * SCH * 4);
  unsigned short* rbfbf = (unsigned short*)alloc((size_t)E3C * NGAUSS * 2);
  float* Cw      = (float*)alloc((size_t)E3C * 4);
  unsigned short* F1bf  = (unsigned short*)alloc((size_t)E3C * SCH * 2);
  float* Wf      = (float*)alloc((size_t)E3C * SCH * 4);
  float* hs      = (float*)alloc((size_t)NN * SCH * 4);
  unsigned short* hsbf  = (unsigned short*)alloc((size_t)NN * SCH * 2);
  float* X1      = (float*)alloc((size_t)NN * SCH * 4);
  float* cat     = (float*)alloc((size_t)NGRAPH * EMB2 * 4);
  unsigned short* catbf = (unsigned short*)alloc((size_t)NGRAPH * EMB2 * 2);
  float* gt      = (float*)alloc((size_t)NGRAPH * 128 * 4);
  float* gbuf    = (float*)alloc((size_t)NGRAPH * 4);
  unsigned short* hfbf  = (unsigned short*)alloc((size_t)NGRAPH * EMB * 2);
  unsigned short* c1bf  = (unsigned short*)alloc((size_t)NGRAPH * 150 * 2);
  (void)ws_size; (void)in_sizes; (void)n_in; (void)out_size;

  // ---- graph counts (inverse, for mean pool) ----
  zerof(stream, cntinv, NGRAPH);
  k_counts<<<blk(NN), 256, 0, stream>>>(batch, cntinv);
  k_invcnt<<<blk(NGRAPH), 256, 0, stream>>>(cntinv);

  // ================= GIN (2D) encoder with virtual node =================
  k_gin_embed<<<blk((long long)NN * EMB), 256, 0, stream>>>(x2d, atom_emb1, atom_emb2, h);
  k_bcast_vn<<<blk((long long)NGRAPH * EMB), 256, 0, stream>>>(vn_init, vn);

  for (int l = 0; l < 5; ++l) {
    k_addvn<<<blk((long long)NN * EMB), 256, 0, stream>>>(h, vn, batch, h_in);
    zerof(stream, agg, (long long)NN * EMB);
    k_gin_msg<<<E2C, 256, 0, stream>>>(src2, dst2, ea2,
                                       gin_bond1 + (size_t)l * 6 * EMB,
                                       gin_bond2 + (size_t)l * 3 * EMB, h_in, agg);
    k_add2bf<<<blk((long long)NN * EMB), 256, 0, stream>>>(h_in, agg, xbfA, (long long)NN * EMB);
    // t1 = relu(X @ W1 + b1)  -> bf16
    tobf(stream, gin_W1 + (size_t)l * EMB * EMB2, wbf, (long long)EMB * EMB2);
    gemm(stream, xbfA, EMB, wbf, EMB2, NN, EMB2, EMB,
         gin_b1 + (size_t)l * EMB2, nullptr, nullptr, nullptr, nullptr, 0, 1,
         nullptr, 0, xbfB, EMB2);
    // t = BN(t1 @ W2 + b2); relu if l<4  -> h (fp32)
    tobf(stream, gin_W2 + (size_t)l * EMB2 * EMB, wbf, (long long)EMB2 * EMB);
    gemm(stream, xbfB, EMB2, wbf, EMB, NN, EMB, EMB2,
         gin_b2 + (size_t)l * EMB, gin_gamma + (size_t)l * EMB, gin_beta + (size_t)l * EMB,
         nullptr, nullptr, 0, (l < 4) ? 1 : 0, h, EMB, nullptr, 0);
    if (l < 4) {  // virtual-node update
      k_copy<<<blk((long long)NGRAPH * EMB), 256, 0, stream>>>(vn, vn_tmp, (long long)NGRAPH * EMB);
      k_segsum<<<blk((long long)NN * EMB), 256, 0, stream>>>(h, batch, vn_tmp, EMB);
      tobf(stream, vn_tmp, vnbfA, (long long)NGRAPH * EMB);
      tobf(stream, vn_W1 + (size_t)l * EMB * EMB2, wbf, (long long)EMB * EMB2);
      gemm(stream, vnbfA, EMB, wbf, EMB2, NGRAPH, EMB2, EMB,
           vn_b1 + (size_t)l * EMB2, nullptr, nullptr, nullptr, nullptr, 0, 1,
           nullptr, 0, vnbfB, EMB2);
      tobf(stream, vn_W2 + (size_t)l * EMB2 * EMB, wbf, (long long)EMB2 * EMB);
      gemm(stream, vnbfB, EMB2, wbf, EMB, NGRAPH, EMB, EMB2,
           vn_b2 + (size_t)l * EMB, nullptr, nullptr, nullptr, nullptr, 0, 1,
           vn, EMB, nullptr, 0);
    }
  }
  zerof(stream, h2d, (long long)NGRAPH * EMB);
  k_segsum<<<blk((long long)NN * EMB), 256, 0, stream>>>(h, batch, h2d, EMB);
  k_rowscale<<<blk((long long)NGRAPH * EMB), 256, 0, stream>>>(h2d, cntinv, NGRAPH, EMB);

  // ================= SchNet (3D) encoder =================
  k_rbf<<<blk(E3C), 256, 0, stream>>>(src3, dst3, pos, rbfbf, Cw);
  k_sch_embed<<<blk((long long)NN * SCH), 256, 0, stream>>>(z, sch_emb, hs, hsbf);

  for (int l = 0; l < 6; ++l) {
    // filter net: F1 = ssp(rbf @ mW1 + mb1)  -> bf16
    tobf(stream, s_mW1 + (size_t)l * NGAUSS * SCH, wbf, (long long)NGAUSS * SCH);
    gemm(stream, rbfbf, NGAUSS, wbf, SCH, E3C, SCH, NGAUSS,
         s_mb1 + (size_t)l * SCH, nullptr, nullptr, nullptr, nullptr, 0, 2,
         nullptr, 0, F1bf, SCH);
    // Wf = (F1 @ mW2 + mb2) * C[:,None]  -> fp32
    tobf(stream, s_mW2 + (size_t)l * SCH * SCH, wbf, (long long)SCH * SCH);
    gemm(stream, F1bf, SCH, wbf, SCH, E3C, SCH, SCH,
         s_mb2 + (size_t)l * SCH, nullptr, nullptr, Cw, nullptr, 0, 0,
         Wf, SCH, nullptr, 0);
    // x1 = hs @ lin1W (no bias)
    tobf(stream, s_lin1W + (size_t)l * SCH * SCH, wbf, (long long)SCH * SCH);
    gemm(stream, hsbf, SCH, wbf, SCH, NN, SCH, SCH,
         nullptr, nullptr, nullptr, nullptr, nullptr, 0, 0, X1, SCH, nullptr, 0);
    // agg = scatter_add( x1[src] * Wf )
    zerof(stream, agg, (long long)NN * SCH);
    k_cfconv<<<E3C, SCH, 0, stream>>>(src3, dst3, X1, Wf, agg);
    k_f32bf<<<blk((long long)NN * SCH), 256, 0, stream>>>(agg, xbfA, (long long)NN * SCH);
    // x2 = ssp(agg @ lin2W + b2) -> bf16
    tobf(stream, s_lin2W + (size_t)l * SCH * SCH, wbf, (long long)SCH * SCH);
    gemm(stream, xbfA, SCH, wbf, SCH, NN, SCH, SCH,
         s_lin2b + (size_t)l * SCH, nullptr, nullptr, nullptr, nullptr, 0, 2,
         nullptr, 0, xbfB, SCH);
    // hs = hs + (x2 @ linW + linb)   (residual; write fp32 + bf16)
    tobf(stream, s_linW + (size_t)l * SCH * SCH, wbf, (long long)SCH * SCH);
    gemm(stream, xbfB, SCH, wbf, SCH, NN, SCH, SCH,
         s_linb + (size_t)l * SCH, nullptr, nullptr, nullptr, hs, SCH, 0,
         hs, SCH, hsbf, SCH);
  }
  zerof(stream, h3d, (long long)NGRAPH * SCH);
  k_segsum<<<blk((long long)NN * SCH), 256, 0, stream>>>(hs, batch, h3d, SCH);
  k_rowscale<<<blk((long long)NGRAPH * SCH), 256, 0, stream>>>(h3d, cntinv, NGRAPH, SCH);

  // ================= gated fusion + classifier =================
  // h2p -> cat[:, 0:300]
  tobf(stream, h2d, vnbfA, (long long)NGRAPH * EMB);
  tobf(stream, p2W, wbf, (long long)EMB * EMB);
  gemm(stream, vnbfA, EMB, wbf, EMB, NGRAPH, EMB, EMB,
       p2b, nullptr, nullptr, nullptr, nullptr, 0, 0, cat, EMB2, nullptr, 0);
  // h3p -> cat[:, 300:600]
  tobf(stream, h3d, vnbfA, (long long)NGRAPH * SCH);
  tobf(stream, p3W, wbf, (long long)SCH * EMB);
  gemm(stream, vnbfA, SCH, wbf, EMB, NGRAPH, EMB, SCH,
       p3b, nullptr, nullptr, nullptr, nullptr, 0, 0, cat + EMB, EMB2, nullptr, 0);
  // gate MLP
  tobf(stream, cat, catbf, (long long)NGRAPH * EMB2);
  tobf(stream, gW1, wbf, (long long)EMB2 * 128);
  gemm(stream, catbf, EMB2, wbf, 128, NGRAPH, 128, EMB2,
       gb1, nullptr, nullptr, nullptr, nullptr, 0, 1, gt, 128, nullptr, 0);
  k_gate<<<blk(NGRAPH), 256, 0, stream>>>(gt, gW2, gb2, gbuf);
  k_fuse<<<blk((long long)NGRAPH * EMB), 256, 0, stream>>>(gbuf, cat, hfbf);
  // classifier
  tobf(stream, cW1, wbf, (long long)EMB * 150);
  gemm(stream, hfbf, EMB, wbf, 150, NGRAPH, 150, EMB,
       cb1, nullptr, nullptr, nullptr, nullptr, 0, 1, nullptr, 0, c1bf, 150);
  tobf(stream, cW2, wbf, (long long)150 * 12);
  gemm(stream, c1bf, 150, wbf, 12, NGRAPH, 12, 150,
       cb2, nullptr, nullptr, nullptr, nullptr, 0, 0, (float*)d_out, 12, nullptr, 0);
}
